// BidirectionalCrossAttention_10986526343406
// MI455X (gfx1250) — matrix operand order repaired and checked by reading, not compiled, not run on hardware
//
#include <hip/hip_runtime.h>
#include <hip/hip_bf16.h>
#include <stddef.h>

// ---------------------------------------------------------------------------
// Bidirectional cross attention for MI455X (gfx1250), all matmuls on
// v_wmma_f32_16x16x32_bf16 (fp32->bf16 inputs, fp32 accumulation).
// B=4, N=1024 (both sides), DIM=1024, H=16, DH=64, SCALE=1/8.
// ---------------------------------------------------------------------------

#define BATCH 4
#define NSEQ  1024      // N_X == N_CTX
#define DIM   1024      // DIM == CTX_DIM == INNER
#define NHEAD 16
#define DH    64
#define SCALE 0.125f

typedef __bf16 v16bf __attribute__((ext_vector_type(16)));
typedef __bf16 v8bf  __attribute__((ext_vector_type(8)));
typedef float  v8f   __attribute__((ext_vector_type(8)));

static __device__ __forceinline__ v8f zero8() {
    v8f z = {0.f,0.f,0.f,0.f,0.f,0.f,0.f,0.f};
    return z;
}

static __device__ __forceinline__ v8f wmma_bf16(v16bf a, v16bf b, v8f c) {
    // D = A(16x32 bf16) x B(32x16 bf16) + C(16x16 f32)
    return __builtin_amdgcn_wmma_f32_16x16x32_bf16(
        /*neg_a=*/false, a, /*neg_b=*/false, b,
        /*c_mod=*/(short)0, c, /*reuse_a=*/false, /*reuse_b=*/false);
}

// A-fragment (16x32, row-major source, leading dim ld):
// lane: m = lane&15, half = lane>>4 ; k(e) = (e<8?0:16) + half*8 + (e&7)
static __device__ __forceinline__ v16bf ldA(const __bf16* base, int ld, int lane) {
    const int m = lane & 15, h = lane >> 4;
    const __bf16* p = base + (size_t)m * ld + h * 8;
    v16bf a;
#pragma unroll
    for (int e = 0; e < 8; ++e) a[e] = p[e];
#pragma unroll
    for (int e = 0; e < 8; ++e) a[8 + e] = p[16 + e];
    return a;
}

// B-fragment (32x16): source laid out "n-major": b[e] = base[n*ld + k(e)], k(e) = (e<8?0:16) + half*8 + (e&7)
// i.e. for each output column n, the 32 k-values are contiguous.
static __device__ __forceinline__ v16bf ldB(const __bf16* base, int ld, int lane) {
    const int n = lane & 15, h = lane >> 4;
    const __bf16* p = base + (size_t)n * ld + h * 8;
    v16bf b;
#pragma unroll
    for (int e = 0; e < 8; ++e) b[e] = p[e];
#pragma unroll
    for (int e = 0; e < 8; ++e) b[8 + e] = p[16 + e];
    return b;
}

// ---------------------------------------------------------------------------
// fp32 -> bf16 conversion
// ---------------------------------------------------------------------------
__global__ void k_f32_to_bf16(const float* __restrict__ in, __bf16* __restrict__ out, int n) {
    for (int i = blockIdx.x * blockDim.x + threadIdx.x; i < n; i += gridDim.x * blockDim.x)
        out[i] = (__bf16)in[i];
}

// ---------------------------------------------------------------------------
// Generic bf16 GEMM:  C[M,N] = A[M,K] * W[K,N] (+bias).  A,W row-major bf16.
// Block: 128 threads = 4 waves, tile 128(M) x 64(N), K-step 64.
// Each wave owns 32 M-rows (two A-fragments); 16 WMMAs per barrier interval.
// W tile (64x64) staged transposed in LDS (b128 loads + b16 scatter), with
// global_prefetch of the next K-tile while WMMAs run.
// ---------------------------------------------------------------------------
__global__ void __launch_bounds__(128)
k_gemm_bf16(const __bf16* __restrict__ A, const __bf16* __restrict__ W,
            const float* __restrict__ bias,
            float* __restrict__ outF, __bf16* __restrict__ outB,
            int M, int N, int K) {
    __shared__ __bf16 WT[64 * 72];              // WT[n][k], padded stride 72
    const int lane = threadIdx.x & 31;
    const int wave = threadIdx.x >> 5;          // 0..3
    const int m0 = blockIdx.x * 128 + wave * 32;
    const int n0 = blockIdx.y * 64;

    v8f acc0[4], acc1[4];
#pragma unroll
    for (int t = 0; t < 4; ++t) { acc0[t] = zero8(); acc1[t] = zero8(); }

    for (int k0 = 0; k0 < K; k0 += 64) {
        __syncthreads();
        // stage W[k0:k0+64, n0:n0+64] transposed: 512 v8 chunks, 4 per thread
#pragma unroll
        for (int c0 = 0; c0 < 4; ++c0) {
            const int c = threadIdx.x + c0 * 128;
            const int n8 = (c & 7) * 8, k = c >> 3;
            const v8bf w = *(const v8bf*)(W + (size_t)(k0 + k) * N + n0 + n8);
#pragma unroll
            for (int e = 0; e < 8; ++e) WT[(n8 + e) * 72 + k] = w[e];
            if (k0 + 64 < K)            // prefetch next K-tile -> global_prefetch_b8
                __builtin_prefetch(W + (size_t)(k0 + 64 + k) * N + n0 + n8, 0, 0);
        }
        __syncthreads();
#pragma unroll
        for (int kk = 0; kk < 64; kk += 32) {
            const v16bf a0 = ldA(A + (size_t)m0 * K + k0 + kk, K, lane);
            const v16bf a1 = ldA(A + (size_t)(m0 + 16) * K + k0 + kk, K, lane);
#pragma unroll
            for (int t = 0; t < 4; ++t) {
                const v16bf b = ldB(WT + t * 16 * 72 + kk, 72, lane);
                acc0[t] = wmma_bf16(a0, b, acc0[t]);
                acc1[t] = wmma_bf16(a1, b, acc1[t]);
            }
        }
    }

    const int cn = lane & 15, ch = lane >> 4;
#pragma unroll
    for (int t = 0; t < 4; ++t) {
        const int gn = n0 + t * 16 + cn;
        const float bv = bias ? bias[gn] : 0.f;
#pragma unroll
        for (int r = 0; r < 8; ++r) {
            const int gm = m0 + r + 8 * ch;
            const float v0 = acc0[t][r] + bv;
            const float v1 = acc1[t][r] + bv;
            if (outF) {
                outF[(size_t)gm * N + gn] = v0;
                outF[(size_t)(gm + 16) * N + gn] = v1;
            } else {
                outB[(size_t)gm * N + gn] = (__bf16)v0;
                outB[(size_t)(gm + 16) * N + gn] = (__bf16)v1;
            }
        }
    }
}

// ---------------------------------------------------------------------------
// Softmax statistics: rowsum[b,h,i] = sum_j exp(sim*s), colsum[b,h,j] = sum_i.
// One block per (b,h); 8 waves, each wave owns i-tiles (stride 8).
// sim tiles recomputed with WMMA straight from global bf16 qk/cqk.
// ---------------------------------------------------------------------------
__global__ void __launch_bounds__(256)
k_stats(const __bf16* __restrict__ qk, const __bf16* __restrict__ cqk,
        float* __restrict__ rowsum, float* __restrict__ colsum) {
    __shared__ float cs[NSEQ];
    const int bh = blockIdx.x, b = bh >> 4, h = bh & 15;
    const int lane = threadIdx.x & 31, wave = threadIdx.x >> 5;
    const int cn = lane & 15, ch = lane >> 4;

    for (int i = threadIdx.x; i < NSEQ; i += blockDim.x) cs[i] = 0.f;
    __syncthreads();

    const __bf16* qbase = qk  + (size_t)b * NSEQ * DIM + h * DH;
    const __bf16* cbase = cqk + (size_t)b * NSEQ * DIM + h * DH;

    for (int it = wave; it < NSEQ / 16; it += 8) {
        const int i0 = it * 16;
        float rowpart[8];
#pragma unroll
        for (int r = 0; r < 8; ++r) rowpart[r] = 0.f;

        for (int jt = 0; jt < NSEQ / 16; ++jt) {
            const int j0 = jt * 16;
            v8f c = zero8();
#pragma unroll
            for (int d0 = 0; d0 < DH; d0 += 32) {
                const v16bf a = ldA(qbase + (size_t)i0 * DIM + d0, DIM, lane);
                const v16bf bb = ldB(cbase + (size_t)j0 * DIM + d0, DIM, lane);
                c = wmma_bf16(a, bb, c);
            }
            float csum = 0.f;
#pragma unroll
            for (int r = 0; r < 8; ++r) {
                const float e = __expf(c[r] * SCALE);
                rowpart[r] += e;
                csum += e;
            }
            atomicAdd(&cs[j0 + cn], csum);          // ds_add_f32
        }
        // reduce rowpart over the 16 lanes of each half (cols n), xor shuffle
#pragma unroll
        for (int r = 0; r < 8; ++r) {
#pragma unroll
            for (int mask = 1; mask < 16; mask <<= 1)
                rowpart[r] += __shfl_xor(rowpart[r], mask, 32);
        }
        if (cn == 0) {
#pragma unroll
            for (int r = 0; r < 8; ++r)
                rowsum[(size_t)bh * NSEQ + i0 + r + 8 * ch] = rowpart[r];
        }
    }
    __syncthreads();
    for (int i = threadIdx.x; i < NSEQ; i += blockDim.x)
        colsum[(size_t)bh * NSEQ + i] = cs[i];
}

// ---------------------------------------------------------------------------
// out[b,h,i,d] = sum_j softmax_row(sim)[i,j] * cv[b,h,j,d]    (flash-style)
// Grid (B*H, 8), 256 threads = 8 waves; each wave owns one i-tile (16 rows).
// Per j-step of 32: recompute sim strip with WMMA, P=exp*invrow -> LDS (bf16,
// A-frag layout), cv tile staged transposed in LDS, then 4 WMMAs (d-tiles).
// ---------------------------------------------------------------------------
__global__ void __launch_bounds__(256)
k_attn_out(const __bf16* __restrict__ qk, const __bf16* __restrict__ cqk,
           const __bf16* __restrict__ cv, const float* __restrict__ rowsum,
           __bf16* __restrict__ outh) {
    __shared__ __bf16 cvT[64 * 40];             // cvT[d][k]
    __shared__ __bf16 stag[8][16 * 40];         // per-wave P tile 16x32
    const int bh = blockIdx.x, b = bh >> 4, h = bh & 15;
    const int lane = threadIdx.x & 31, wave = threadIdx.x >> 5;
    const int i0 = (blockIdx.y * 8 + wave) * 16;
    const int cn = lane & 15, ch = lane >> 4;

    const __bf16* qbase = qk  + (size_t)b * NSEQ * DIM + h * DH;
    const __bf16* cbase = cqk + (size_t)b * NSEQ * DIM + h * DH;
    const __bf16* vbase = cv  + (size_t)b * NSEQ * DIM + h * DH;

    float invrow[8];
#pragma unroll
    for (int r = 0; r < 8; ++r)
        invrow[r] = 1.0f / rowsum[(size_t)bh * NSEQ + i0 + r + 8 * ch];

    // staging indices for the transposed cv tile (one v8 chunk per thread)
    const int fd8 = (threadIdx.x & 7) * 8, fk = threadIdx.x >> 3;

    v8f acc[4];
#pragma unroll
    for (int t = 0; t < 4; ++t) acc[t] = zero8();

    for (int j0 = 0; j0 < NSEQ; j0 += 32) {
        __syncthreads();
        {
            const v8bf w = *(const v8bf*)(vbase + (size_t)(j0 + fk) * DIM + fd8);
#pragma unroll
            for (int e = 0; e < 8; ++e) cvT[(fd8 + e) * 40 + fk] = w[e];
        }
        __syncthreads();

        __bf16* st = stag[wave];
#pragma unroll
        for (int js = 0; js < 2; ++js) {
            v8f c = zero8();
#pragma unroll
            for (int d0 = 0; d0 < DH; d0 += 32) {
                const v16bf a = ldA(qbase + (size_t)i0 * DIM + d0, DIM, lane);
                const v16bf bb = ldB(cbase + (size_t)(j0 + js * 16) * DIM + d0, DIM, lane);
                c = wmma_bf16(a, bb, c);
            }
#pragma unroll
            for (int r = 0; r < 8; ++r) {
                const float p = __expf(c[r] * SCALE) * invrow[r];
                st[(r + 8 * ch) * 40 + js * 16 + cn] = (__bf16)p;
            }
        }
        const v16bf pa = ldA(st, 40, lane);     // P as A-fragment (K=32 over j)
#pragma unroll
        for (int t = 0; t < 4; ++t) {
            const v16bf bb = ldB(cvT + t * 16 * 40, 40, lane);
            acc[t] = wmma_bf16(pa, bb, acc[t]);
        }
    }
#pragma unroll
    for (int t = 0; t < 4; ++t)
#pragma unroll
        for (int r = 0; r < 8; ++r)
            outh[((size_t)b * NSEQ + i0 + r + 8 * ch) * DIM + h * DH + t * 16 + cn] =
                (__bf16)acc[t][r];
}

// ---------------------------------------------------------------------------
// context_out[b,h,j,d] = sum_i softmax_col(sim)[i,j] * v[b,h,i,d]
// Mirror of k_attn_out: waves own j-tiles, loop over i, sim computed
// transposed (A = cqk rows j, B = qk rows i), divide by colsum[j].
// ---------------------------------------------------------------------------
__global__ void __launch_bounds__(256)
k_attn_ctx(const __bf16* __restrict__ qk, const __bf16* __restrict__ cqk,
           const __bf16* __restrict__ v, const float* __restrict__ colsum,
           __bf16* __restrict__ ctxh) {
    __shared__ __bf16 vT[64 * 40];              // vT[d][k]
    __shared__ __bf16 stag[8][16 * 40];         // per-wave P'^T tile 16x32
    const int bh = blockIdx.x, b = bh >> 4, h = bh & 15;
    const int lane = threadIdx.x & 31, wave = threadIdx.x >> 5;
    const int j0 = (blockIdx.y * 8 + wave) * 16;
    const int cn = lane & 15, ch = lane >> 4;

    const __bf16* qbase = qk  + (size_t)b * NSEQ * DIM + h * DH;
    const __bf16* cbase = cqk + (size_t)b * NSEQ * DIM + h * DH;
    const __bf16* vbase = v   + (size_t)b * NSEQ * DIM + h * DH;

    float invcol[8];
#pragma unroll
    for (int r = 0; r < 8; ++r)
        invcol[r] = 1.0f / colsum[(size_t)bh * NSEQ + j0 + r + 8 * ch];

    const int fd8 = (threadIdx.x & 7) * 8, fk = threadIdx.x >> 3;

    v8f acc[4];
#pragma unroll
    for (int t = 0; t < 4; ++t) acc[t] = zero8();

    for (int i0 = 0; i0 < NSEQ; i0 += 32) {
        __syncthreads();
        {
            const v8bf w = *(const v8bf*)(vbase + (size_t)(i0 + fk) * DIM + fd8);
#pragma unroll
            for (int e = 0; e < 8; ++e) vT[(fd8 + e) * 40 + fk] = w[e];
        }
        __syncthreads();

        __bf16* st = stag[wave];
#pragma unroll
        for (int is = 0; is < 2; ++is) {
            v8f c = zero8();                    // simT tile: rows j, cols i
#pragma unroll
            for (int d0 = 0; d0 < DH; d0 += 32) {
                const v16bf a = ldA(cbase + (size_t)j0 * DIM + d0, DIM, lane);
                const v16bf bb = ldB(qbase + (size_t)(i0 + is * 16) * DIM + d0, DIM, lane);
                c = wmma_bf16(a, bb, c);
            }
#pragma unroll
            for (int r = 0; r < 8; ++r) {
                const float p = __expf(c[r] * SCALE) * invcol[r];
                st[(r + 8 * ch) * 40 + is * 16 + cn] = (__bf16)p;
            }
        }
        const v16bf pa = ldA(st, 40, lane);     // P'^T as A-fragment (K=32 over i)
#pragma unroll
        for (int t = 0; t < 4; ++t) {
            const v16bf bb = ldB(vT + t * 16 * 40, 40, lane);
            acc[t] = wmma_bf16(pa, bb, acc[t]);
        }
    }
#pragma unroll
    for (int t = 0; t < 4; ++t)
#pragma unroll
        for (int r = 0; r < 8; ++r)
            ctxh[((size_t)b * NSEQ + j0 + r + 8 * ch) * DIM + h * DH + t * 16 + cn] =
                (__bf16)acc[t][r];
}

// ---------------------------------------------------------------------------
// Host orchestration
// ---------------------------------------------------------------------------
extern "C" void kernel_launch(void* const* d_in, const int* in_sizes, int n_in,
                              void* d_out, int out_size, void* d_ws, size_t ws_size,
                              hipStream_t stream) {
    const float* x      = (const float*)d_in[0];
    const float* ctx    = (const float*)d_in[1];
    const float* w_qk   = (const float*)d_in[2];
    const float* w_v    = (const float*)d_in[3];
    const float* cw_qk  = (const float*)d_in[4];
    const float* cw_v   = (const float*)d_in[5];
    const float* w_out  = (const float*)d_in[6];
    const float* b_out  = (const float*)d_in[7];
    const float* cw_out = (const float*)d_in[8];
    const float* cb_out = (const float*)d_in[9];
    float* out = (float*)d_out;

    const size_t ACT  = (size_t)BATCH * NSEQ * DIM;   // 4M elems
    const size_t WEL  = (size_t)DIM * DIM;            // 1M elems
    const size_t SUMS = (size_t)BATCH * NHEAD * NSEQ; // 64K elems

    char* p = (char*)d_ws;
    size_t off = 0;
    auto carve = [&](size_t bytes) { void* q = p + off; off = (off + bytes + 255) & ~(size_t)255; return q; };
    __bf16* xb     = (__bf16*)carve(ACT * 2);
    __bf16* ctxb   = (__bf16*)carve(ACT * 2);
    __bf16* wqkb   = (__bf16*)carve(WEL * 2);
    __bf16* wvb    = (__bf16*)carve(WEL * 2);
    __bf16* cwqkb  = (__bf16*)carve(WEL * 2);
    __bf16* cwvb   = (__bf16*)carve(WEL * 2);
    __bf16* woutb  = (__bf16*)carve(WEL * 2);
    __bf16* cwoutb = (__bf16*)carve(WEL * 2);
    __bf16* qkb    = (__bf16*)carve(ACT * 2);
    __bf16* vb     = (__bf16*)carve(ACT * 2);
    __bf16* cqkb   = (__bf16*)carve(ACT * 2);
    __bf16* cvb    = (__bf16*)carve(ACT * 2);
    __bf16* outhb  = (__bf16*)carve(ACT * 2);
    __bf16* ctxhb  = (__bf16*)carve(ACT * 2);
    float*  rowsum = (float*)carve(SUMS * 4);
    float*  colsum = (float*)carve(SUMS * 4);
    (void)ws_size; (void)in_sizes; (void)n_in; (void)out_size;

    // 1) fp32 -> bf16 conversions
    k_f32_to_bf16<<<1024, 256, 0, stream>>>(x,      xb,     (int)ACT);
    k_f32_to_bf16<<<1024, 256, 0, stream>>>(ctx,    ctxb,   (int)ACT);
    k_f32_to_bf16<<<256,  256, 0, stream>>>(w_qk,   wqkb,   (int)WEL);
    k_f32_to_bf16<<<256,  256, 0, stream>>>(w_v,    wvb,    (int)WEL);
    k_f32_to_bf16<<<256,  256, 0, stream>>>(cw_qk,  cwqkb,  (int)WEL);
    k_f32_to_bf16<<<256,  256, 0, stream>>>(cw_v,   cwvb,   (int)WEL);
    k_f32_to_bf16<<<256,  256, 0, stream>>>(w_out,  woutb,  (int)WEL);
    k_f32_to_bf16<<<256,  256, 0, stream>>>(cw_out, cwoutb, (int)WEL);

    // 2) projection GEMMs (bf16 out), M = B*N = 4096, N = K = 1024
    const int M = BATCH * NSEQ;
    dim3 ggrid(M / 128, DIM / 64);
    k_gemm_bf16<<<ggrid, 128, 0, stream>>>(xb,   wqkb,  nullptr, nullptr, qkb,  M, DIM, DIM);
    k_gemm_bf16<<<ggrid, 128, 0, stream>>>(xb,   wvb,   nullptr, nullptr, vb,   M, DIM, DIM);
    k_gemm_bf16<<<ggrid, 128, 0, stream>>>(ctxb, cwqkb, nullptr, nullptr, cqkb, M, DIM, DIM);
    k_gemm_bf16<<<ggrid, 128, 0, stream>>>(ctxb, cwvb,  nullptr, nullptr, cvb,  M, DIM, DIM);

    // 3) softmax statistics (row & col sums of exp(sim*scale))
    k_stats<<<BATCH * NHEAD, 256, 0, stream>>>(qkb, cqkb, rowsum, colsum);

    // 4) attention passes (recompute sim, WMMA P@cv and P'^T@v)
    dim3 agrid(BATCH * NHEAD, 8);
    k_attn_out<<<agrid, 256, 0, stream>>>(qkb, cqkb, cvb, rowsum, outhb);
    k_attn_ctx<<<agrid, 256, 0, stream>>>(qkb, cqkb, vb,  colsum, ctxhb);

    // 5) output projections with bias, fp32 into d_out (out then context_out)
    k_gemm_bf16<<<ggrid, 128, 0, stream>>>(outhb, woutb,  b_out,  out,       nullptr, M, DIM, DIM);
    k_gemm_bf16<<<ggrid, 128, 0, stream>>>(ctxhb, cwoutb, cb_out, out + ACT, nullptr, M, DIM, DIM);
}
